// MAEDecoder_85109071937962
// MI455X (gfx1250) — compile-verified, matrix operand order checked by reading, not run on hardware
//
#include <hip/hip_runtime.h>
#include <hip/hip_bf16.h>
#include <math.h>
#include <stdint.h>

typedef __attribute__((ext_vector_type(16))) _Float16 v16h;
typedef __attribute__((ext_vector_type(8)))  _Float16 v8h;
typedef __attribute__((ext_vector_type(8)))  float    v8f;

#define DEV __device__ __forceinline__

#define BB   16
#define PER  2048
#define VISN 512
#define MSKN 1536
#define DD   256
#define HIDN 128
#define NHD  8
#define DHD  32
#define NTOT (BB*PER)    // 32768
#define NMSK (BB*MSKN)   // 24576
#define NVIS (BB*VISN)   // 8192

// ---- helpers -------------------------------------------------------------

DEV v16h loadfrag(const _Float16* p) {
  // A/B fragment for v_wmma_f32_16x16x32_f16 (wave32):
  // lane holds K = hi*8..hi*8+7  and  K = 16+hi*8..16+hi*8+7  (caller adds hi*8)
  v8h a = *(const v8h*)(p);
  v8h b = *(const v8h*)(p + 16);
  return __builtin_shufflevector(a, b, 0,1,2,3,4,5,6,7,8,9,10,11,12,13,14,15);
}

DEV v8f wmma(v16h a, v16h b, v8f c) {
  return __builtin_amdgcn_wmma_f32_16x16x32_f16(false, a, false, b, (short)0, c,
                                                false, false);
}

DEV float geluf(float x) { return 0.5f * x * (1.0f + erff(x * 0.70710678118654752f)); }

// Async global->LDS copy (CDNA5 ASYNCcnt path). ldsoff: byte offset in LDS,
// goff: byte offset added to 64-bit SGPR base. GVS addressing mode.
DEV void async_copy16(uint32_t ldsoff, uint32_t goff, const void* sbase) {
  asm volatile("global_load_async_to_lds_b128 %0, %1, %2"
               :: "v"(ldsoff), "v"(goff), "s"(sbase)
               : "memory");
}
DEV void async_wait0() {
  asm volatile("s_wait_asynccnt 0x0" ::: "memory");
}

// ---- min/max header ------------------------------------------------------

__global__ void k_init_hdr(int* hdr) {
  int t = threadIdx.x;
  if (t < 3)      hdr[t] = 0x7f800000;          // +inf  (mins)
  else if (t < 6) hdr[t] = (int)0xff800000u;    // -inf  (maxs)
}

__global__ void k_minmax(const float* __restrict__ coord, int* hdr) {
  int tid = blockIdx.x * blockDim.x + threadIdx.x;
  int stride = gridDim.x * blockDim.x;
  float mn[3] = { INFINITY, INFINITY, INFINITY };
  float mx[3] = { -INFINITY, -INFINITY, -INFINITY };
  for (int i = tid; i < NTOT; i += stride) {
    #pragma unroll
    for (int d = 0; d < 3; d++) {
      float c = coord[i * 3 + d];
      mn[d] = fminf(mn[d], c);
      mx[d] = fmaxf(mx[d], c);
    }
  }
  #pragma unroll
  for (int off = 16; off >= 1; off >>= 1) {
    #pragma unroll
    for (int d = 0; d < 3; d++) {
      mn[d] = fminf(mn[d], __shfl_xor(mn[d], off, 32));
      mx[d] = fmaxf(mx[d], __shfl_xor(mx[d], off, 32));
    }
  }
  if ((threadIdx.x & 31) == 0) {
    #pragma unroll
    for (int d = 0; d < 3; d++) {
      atomicMin(&hdr[d],     __float_as_int(mn[d]));   // coords >= 0: int order ok
      atomicMax(&hdr[3 + d], __float_as_int(mx[d]));
    }
  }
}

// ---- fp32 -> f16 weight convert -----------------------------------------

__global__ void k_f32tof16(const float* __restrict__ src, _Float16* __restrict__ dst,
                           int n) {
  int i = blockIdx.x * blockDim.x + threadIdx.x;
  if (i < n) dst[i] = (_Float16)src[i];
}

// ---- pos-embed stage 1: cn@W1^T + b1 -> LN -> GELU -> f16 ---------------

__global__ void k_pos1(const float* __restrict__ coord, const int* __restrict__ hdr,
                       const float* __restrict__ w1, const float* __restrict__ b1,
                       const float* __restrict__ g,  const float* __restrict__ be,
                       _Float16* __restrict__ gout) {
  __shared__ float s1[DD], s2[DD];
  int r = blockIdx.x, t = threadIdx.x;
  float cn[3];
  #pragma unroll
  for (int d = 0; d < 3; d++) {
    float cmin = __int_as_float(hdr[d]);
    float cmax = __int_as_float(hdr[3 + d]);
    float span = fmaxf(cmax - cmin, 1.0f);
    cn[d] = (coord[r * 3 + d] - cmin) / span;
  }
  float e = b1[t] + cn[0] * w1[t * 3 + 0] + cn[1] * w1[t * 3 + 1] + cn[2] * w1[t * 3 + 2];
  s1[t] = e; s2[t] = e * e;
  __syncthreads();
  for (int s = DD / 2; s > 0; s >>= 1) {
    if (t < s) { s1[t] += s1[t + s]; s2[t] += s2[t + s]; }
    __syncthreads();
  }
  float m   = s1[0] * (1.0f / DD);
  float var = s2[0] * (1.0f / DD) - m * m;
  float y   = (e - m) * rsqrtf(var + 1e-5f) * g[t] + be[t];
  gout[(size_t)r * DD + t] = (_Float16)geluf(y);
}

// ---- generic WMMA GEMM: out[M,N] = A[M,K] @ W[N,K]^T + bias -------------
// 4 waves per workgroup (64 rows); the 64xK weight strip for this n-group is
// staged once into LDS via async global->LDS copies and shared by all waves.
// MODE 0: fp32 out.  MODE 1: f16 out.  MODE 2: f16 out, V head-transposed.
// A-row remap: global_row = (m/blk)*blkStride + m%blk  (for reading X_dec as kv)

template<int MODE>
DEV void epi(v8f acc, int row0, int col, float bb, void* outp, int N) {
  #pragma unroll
  for (int j = 0; j < 8; j++) {
    int row = row0 + j;
    float v = acc[j] + bb;
    if (MODE == 0) {
      ((float*)outp)[(size_t)row * N + col] = v;
    } else if (MODE == 1) {
      ((_Float16*)outp)[(size_t)row * N + col] = (_Float16)v;
    } else {
      int bidx = row >> 9, vis = row & 511;      // kv row -> (b, vis)
      int hh = col >> 5,  dh = col & 31;         // col   -> (head, dh)
      ((_Float16*)outp)[((size_t)(((bidx << 3) + hh) << 5) + dh) * 512 + vis] =
          (_Float16)v;
    }
  }
}

template<int MODE>
__global__ void __launch_bounds__(128)
k_gemm(const _Float16* __restrict__ A, int lda, int blk, int blkStride,
       const _Float16* __restrict__ W, int ldw,
       const float* __restrict__ bias,
       void* __restrict__ outp, int N, int K) {
  __shared__ __align__(16) _Float16 wlds[64 * DD];   // 32 KB max (K <= 256)
  int t = threadIdx.x;
  int wave = t >> 5;
  int lane = t & 31;
  int lo = lane & 15, hi = lane >> 4;
  int m0 = blockIdx.x * 64 + wave * 16;
  int n0 = blockIdx.y * 64;

  // ---- stage 64 x K weight strip into LDS (async, ASYNCcnt) ----
  {
    const _Float16* wbase = W + (size_t)n0 * ldw;
    uint32_t ldsbase = (uint32_t)(uintptr_t)(&wlds[0]);
    int cpr = K >> 3;                 // 16B chunks per row
    for (int c = t; c < 64 * cpr; c += 128) {
      int r  = c / cpr;
      int cc = c - r * cpr;
      uint32_t ldsoff = ldsbase + (uint32_t)(r * K + cc * 8) * 2u;
      uint32_t goff   = (uint32_t)((r * ldw + cc * 8) * 2);
      async_copy16(ldsoff, goff, (const void*)wbase);
    }
    async_wait0();
  }
  __syncthreads();

  int mrow = m0 + lo;
  size_t arow = (size_t)(mrow / blk) * (size_t)blkStride + (size_t)(mrow % blk);
  const _Float16* ap = A + arow * lda + hi * 8;
  const _Float16* wl = &wlds[(size_t)lo * K + hi * 8];   // strip-relative row lo
  v8f acc0 = {}, acc1 = {}, acc2 = {}, acc3 = {};
  for (int k0 = 0; k0 < K; k0 += 32) {
    v16h a = loadfrag(ap + k0);
    acc0 = wmma(a, loadfrag(wl + k0),          acc0);
    acc1 = wmma(a, loadfrag(wl + 16 * K + k0), acc1);
    acc2 = wmma(a, loadfrag(wl + 32 * K + k0), acc2);
    acc3 = wmma(a, loadfrag(wl + 48 * K + k0), acc3);
  }
  int row0 = m0 + hi * 8;
  epi<MODE>(acc0, row0, n0 +  0 + lo, bias[n0 +  0 + lo], outp, N);
  epi<MODE>(acc1, row0, n0 + 16 + lo, bias[n0 + 16 + lo], outp, N);
  epi<MODE>(acc2, row0, n0 + 32 + lo, bias[n0 + 32 + lo], outp, N);
  epi<MODE>(acc3, row0, n0 + 48 + lo, bias[n0 + 48 + lo], outp, N);
}

// ---- combine pos with encoded / mask token ------------------------------

__global__ void k_comb_vis(const float* __restrict__ enc, const float* __restrict__ pos,
                           _Float16* __restrict__ X) {
  int idx = blockIdx.x * 256 + threadIdx.x;   // NVIS*DD threads
  int r = idx >> 8, d = idx & 255;
  int b = r / VISN, j = r % VISN;
  size_t g = (size_t)(b * PER + j) * DD + d;
  X[g] = (_Float16)(enc[(size_t)r * DD + d] + pos[g]);
}

__global__ void k_comb_msk(const float* __restrict__ mtok, const float* __restrict__ pos,
                           _Float16* __restrict__ q16) {
  int idx = blockIdx.x * 256 + threadIdx.x;   // NMSK*DD threads
  int r = idx >> 8, d = idx & 255;
  int b = r / MSKN, i = r % MSKN;
  size_t g = (size_t)(b * PER + VISN + i) * DD + d;
  q16[(size_t)r * DD + d] = (_Float16)(mtok[d] + pos[g]);
}

// ---- fused attention: one wave per (b, h, 16-row tile) ------------------

__global__ void __launch_bounds__(32)
k_attn(const _Float16* __restrict__ Q, const _Float16* __restrict__ Kp,
       const _Float16* __restrict__ Vt, _Float16* __restrict__ O) {
  __shared__ __align__(16) float    sc[16 * VISN];     // 32 KB
  __shared__ __align__(16) _Float16 pbuf[16 * VISN];   // 16 KB
  int lane = threadIdx.x & 31;
  int lo = lane & 15, hi = lane >> 4;
  int tile = blockIdx.x;
  int mt = tile % (MSKN / 16);
  int h  = (tile / (MSKN / 16)) & (NHD - 1);
  int b  = tile / ((MSKN / 16) * NHD);
  int m0 = mt * 16;
  const float scale = 0.17677669529663687f;  // 1/sqrt(32)

  // scores strip: Qh[16x32] @ Kh^T -> sc[16x512]
  v16h aq = loadfrag(Q + (size_t)(b * MSKN + m0 + lo) * DD + h * DHD + hi * 8);
  for (int n0 = 0; n0 < VISN; n0 += 16) {
    v16h bk = loadfrag(Kp + (size_t)(b * VISN + n0 + lo) * DD + h * DHD + hi * 8);
    v8f c = {};
    c = wmma(aq, bk, c);
    #pragma unroll
    for (int j = 0; j < 8; j++) sc[(hi * 8 + j) * VISN + n0 + lo] = c[j] * scale;
  }
  __syncthreads();

  // softmax over 512 keys, wave-cooperative per row
  for (int m = 0; m < 16; m++) {
    float v[16];
    float mx = -INFINITY;
    #pragma unroll
    for (int i = 0; i < 16; i++) { v[i] = sc[m * VISN + lane + i * 32]; mx = fmaxf(mx, v[i]); }
    #pragma unroll
    for (int off = 16; off >= 1; off >>= 1) mx = fmaxf(mx, __shfl_xor(mx, off, 32));
    float s = 0.0f;
    #pragma unroll
    for (int i = 0; i < 16; i++) { v[i] = __expf(v[i] - mx); s += v[i]; }
    #pragma unroll
    for (int off = 16; off >= 1; off >>= 1) s += __shfl_xor(s, off, 32);
    float inv = 1.0f / s;
    #pragma unroll
    for (int i = 0; i < 16; i++) pbuf[m * VISN + lane + i * 32] = (_Float16)(v[i] * inv);
  }
  __syncthreads();

  // O tile: P[16x512] @ Vh[512x32]  (Vt is head-transposed, rows contiguous)
  v8f o0 = {}, o1 = {};
  const _Float16* vbase = Vt + (size_t)((b * NHD + h) * DHD) * VISN;
  for (int kc = 0; kc < VISN; kc += 32) {
    v16h ap = loadfrag(&pbuf[lo * VISN + kc + hi * 8]);
    o0 = wmma(ap, loadfrag(vbase + (size_t)lo * VISN + kc + hi * 8), o0);
    o1 = wmma(ap, loadfrag(vbase + (size_t)(16 + lo) * VISN + kc + hi * 8), o1);
  }
  #pragma unroll
  for (int j = 0; j < 8; j++) {
    size_t rb = (size_t)(b * MSKN + m0 + hi * 8 + j) * DD + h * DHD;
    O[rb + lo]      = (_Float16)o0[j];
    O[rb + 16 + lo] = (_Float16)o1[j];
  }
}

// ---- masked features: LN(q + attn_out), write into X_dec ----------------

__global__ void k_mfln(const float* __restrict__ attn, const float* __restrict__ pos,
                       const float* __restrict__ mtok, const float* __restrict__ g,
                       const float* __restrict__ be, _Float16* __restrict__ X) {
  __shared__ float s1[DD], s2[DD];
  int r = blockIdx.x, t = threadIdx.x;
  int b = r / MSKN, i = r % MSKN;
  size_t gidx = (size_t)(b * PER + VISN + i) * DD + t;
  float x = mtok[t] + pos[gidx] + attn[(size_t)r * DD + t];
  s1[t] = x; s2[t] = x * x;
  __syncthreads();
  for (int s = DD / 2; s > 0; s >>= 1) {
    if (t < s) { s1[t] += s1[t + s]; s2[t] += s2[t + s]; }
    __syncthreads();
  }
  float m = s1[0] * (1.0f / DD), var = s2[0] * (1.0f / DD) - m * m;
  X[gidx] = (_Float16)((x - m) * rsqrtf(var + 1e-5f) * g[t] + be[t]);
}

// ---- decoder layer-1 post: LN + GELU -> f16 -----------------------------

__global__ void k_d1post(const float* __restrict__ H1, const float* __restrict__ g,
                         const float* __restrict__ be, _Float16* __restrict__ H1h) {
  __shared__ float s1[HIDN], s2[HIDN];
  int r = blockIdx.x, t = threadIdx.x;
  float x = H1[(size_t)r * HIDN + t];
  s1[t] = x; s2[t] = x * x;
  __syncthreads();
  for (int s = HIDN / 2; s > 0; s >>= 1) {
    if (t < s) { s1[t] += s1[t + s]; s2[t] += s2[t + s]; }
    __syncthreads();
  }
  float m = s1[0] * (1.0f / HIDN), var = s2[0] * (1.0f / HIDN) - m * m;
  H1h[(size_t)r * HIDN + t] = (_Float16)geluf((x - m) * rsqrtf(var + 1e-5f) * g[t] + be[t]);
}

// ---- final: GELU(h2) @ w3^T + b3 -> recon -------------------------------

__global__ void k_final(const float* __restrict__ H2, const float* __restrict__ w3,
                        const float* __restrict__ b3, float* __restrict__ out) {
  __shared__ float w3s[4 * HIDN];
  for (int i = threadIdx.x; i < 4 * HIDN; i += blockDim.x) w3s[i] = w3[i];
  __syncthreads();
  int r = blockIdx.x * blockDim.x + threadIdx.x;
  float a0 = b3[0], a1 = b3[1], a2 = b3[2], a3 = b3[3];
  for (int k = 0; k < HIDN; k++) {
    float x = geluf(H2[(size_t)r * HIDN + k]);
    a0 += x * w3s[k];
    a1 += x * w3s[HIDN + k];
    a2 += x * w3s[2 * HIDN + k];
    a3 += x * w3s[3 * HIDN + k];
  }
  out[(size_t)r * 4 + 0] = a0;
  out[(size_t)r * 4 + 1] = a1;
  out[(size_t)r * 4 + 2] = a2;
  out[(size_t)r * 4 + 3] = a3;
}

// ---- host orchestration --------------------------------------------------

extern "C" void kernel_launch(void* const* d_in, const int* in_sizes, int n_in,
                              void* d_out, int out_size, void* d_ws, size_t ws_size,
                              hipStream_t stream) {
  const float* encoded = (const float*)d_in[0];
  const float* coord   = (const float*)d_in[1];
  const float* mtok    = (const float*)d_in[2];
  const float* pe_w1   = (const float*)d_in[3];
  const float* pe_b1   = (const float*)d_in[4];
  const float* pe_g    = (const float*)d_in[5];
  const float* pe_be   = (const float*)d_in[6];
  const float* pe_w2   = (const float*)d_in[7];
  const float* pe_b2   = (const float*)d_in[8];
  const float* wq      = (const float*)d_in[9];
  const float* bq      = (const float*)d_in[10];
  const float* wk      = (const float*)d_in[11];
  const float* bk      = (const float*)d_in[12];
  const float* wv      = (const float*)d_in[13];
  const float* bv      = (const float*)d_in[14];
  const float* wo      = (const float*)d_in[15];
  const float* bo      = (const float*)d_in[16];
  const float* an_g    = (const float*)d_in[17];
  const float* an_b    = (const float*)d_in[18];
  const float* d_w1    = (const float*)d_in[19];
  const float* d_b1    = (const float*)d_in[20];
  const float* dg      = (const float*)d_in[21];
  const float* dbe     = (const float*)d_in[22];
  const float* d_w2    = (const float*)d_in[23];
  const float* d_b2    = (const float*)d_in[24];
  const float* d_w3    = (const float*)d_in[25];
  const float* d_b3    = (const float*)d_in[26];
  float* recon = (float*)d_out;

  char* ws = (char*)d_ws;
  size_t off = 256;  // header: 6 ints for coord min/max
  auto take = [&](size_t bytes) {
    size_t o = off;
    off = (off + bytes + 255) & ~(size_t)255;
    return o;
  };

  // persistent f16 weights
  size_t o_pw2 = take((size_t)DD * DD * 2);
  size_t o_wq  = take((size_t)DD * DD * 2);
  size_t o_wk  = take((size_t)DD * DD * 2);
  size_t o_wv  = take((size_t)DD * DD * 2);
  size_t o_wo  = take((size_t)DD * DD * 2);
  size_t o_dw1 = take((size_t)HIDN * DD * 2);
  size_t o_dw2 = take((size_t)HIDN * HIDN * 2);
  // region: g (pos-embed hidden, f16) — later reused for attention output O
  size_t o_g = take((size_t)NTOT * DD * 2);   // 16 MB
  size_t o_O = o_g;                            // 12 MB alias (g dead after pos GEMM)
  // region: q_f16 + Qp — later reused for attn_out fp32 (24 MB)
  size_t o_q16  = take((size_t)NMSK * DD * 2); // 12 MB
  size_t o_Qp   = take((size_t)NMSK * DD * 2); // 12 MB
  size_t o_attn = o_q16;                       // alias
  // region: pos fp32 + Kp + Vt (contiguous 40 MB) — later decoder hiddens
  size_t o_pos = take((size_t)NTOT * DD * 4);  // 32 MB
  size_t o_Kp  = take((size_t)NVIS * DD * 2);  // 4 MB
  size_t o_Vt  = take((size_t)NVIS * DD * 2);  // 4 MB
  size_t o_H1  = o_pos;                                              // 16 MB
  size_t o_H1h = o_pos + (size_t)NTOT * HIDN * 4;                    //  8 MB
  size_t o_H2  = o_pos + (size_t)NTOT * HIDN * 4 + (size_t)NTOT * HIDN * 2; // 16 MB
  // decoder input (kv rows + masked-feature rows in global point order)
  size_t o_X = take((size_t)NTOT * DD * 2);    // 16 MB

  int* hdr = (int*)ws;
  const int IDBLK = 1 << 28;

  // 1) coord min/max
  k_init_hdr<<<1, 32, 0, stream>>>(hdr);
  k_minmax<<<64, 256, 0, stream>>>(coord, hdr);

  // 2) weights -> f16
  k_f32tof16<<<DD * DD / 256, 256, 0, stream>>>(pe_w2, (_Float16*)(ws + o_pw2), DD * DD);
  k_f32tof16<<<DD * DD / 256, 256, 0, stream>>>(wq, (_Float16*)(ws + o_wq), DD * DD);
  k_f32tof16<<<DD * DD / 256, 256, 0, stream>>>(wk, (_Float16*)(ws + o_wk), DD * DD);
  k_f32tof16<<<DD * DD / 256, 256, 0, stream>>>(wv, (_Float16*)(ws + o_wv), DD * DD);
  k_f32tof16<<<DD * DD / 256, 256, 0, stream>>>(wo, (_Float16*)(ws + o_wo), DD * DD);
  k_f32tof16<<<HIDN * DD / 256, 256, 0, stream>>>(d_w1, (_Float16*)(ws + o_dw1), HIDN * DD);
  k_f32tof16<<<HIDN * HIDN / 256, 256, 0, stream>>>(d_w2, (_Float16*)(ws + o_dw2), HIDN * HIDN);

  // 3) pos-embed stage 1 -> g (f16)
  k_pos1<<<NTOT, DD, 0, stream>>>(coord, hdr, pe_w1, pe_b1, pe_g, pe_be,
                                  (_Float16*)(ws + o_g));
  // 4) pos = g @ pe_w2^T + pe_b2 (fp32)
  k_gemm<0><<<dim3(NTOT / 64, DD / 64), 128, 0, stream>>>(
      (const _Float16*)(ws + o_g), DD, IDBLK, 0,
      (const _Float16*)(ws + o_pw2), DD, pe_b2, ws + o_pos, DD, DD);

  // 5) kv -> X_dec (visible rows, f16); q (f16)
  k_comb_vis<<<NVIS * DD / 256, 256, 0, stream>>>(encoded, (const float*)(ws + o_pos),
                                                  (_Float16*)(ws + o_X));
  k_comb_msk<<<NMSK * DD / 256, 256, 0, stream>>>(mtok, (const float*)(ws + o_pos),
                                                  (_Float16*)(ws + o_q16));

  // 6) projections (Q, K f16; V head-transposed f16)
  k_gemm<1><<<dim3(NMSK / 64, DD / 64), 128, 0, stream>>>(
      (const _Float16*)(ws + o_q16), DD, IDBLK, 0,
      (const _Float16*)(ws + o_wq), DD, bq, ws + o_Qp, DD, DD);
  k_gemm<1><<<dim3(NVIS / 64, DD / 64), 128, 0, stream>>>(
      (const _Float16*)(ws + o_X), DD, VISN, PER,
      (const _Float16*)(ws + o_wk), DD, bk, ws + o_Kp, DD, DD);
  k_gemm<2><<<dim3(NVIS / 64, DD / 64), 128, 0, stream>>>(
      (const _Float16*)(ws + o_X), DD, VISN, PER,
      (const _Float16*)(ws + o_wv), DD, bv, ws + o_Vt, DD, DD);

  // 7) fused attention -> O (f16, reuses g region)
  k_attn<<<BB * NHD * (MSKN / 16), 32, 0, stream>>>(
      (const _Float16*)(ws + o_Qp), (const _Float16*)(ws + o_Kp),
      (const _Float16*)(ws + o_Vt), (_Float16*)(ws + o_O));

  // 8) attn_out = O @ wo^T + bo (fp32, reuses q16+Qp region)
  k_gemm<0><<<dim3(NMSK / 64, DD / 64), 128, 0, stream>>>(
      (const _Float16*)(ws + o_O), DD, IDBLK, 0,
      (const _Float16*)(ws + o_wo), DD, bo, ws + o_attn, DD, DD);

  // 9) masked_features = LN(q + attn_out) -> X_dec masked rows (f16)
  k_mfln<<<NMSK, DD, 0, stream>>>((const float*)(ws + o_attn), (const float*)(ws + o_pos),
                                  mtok, an_g, an_b, (_Float16*)(ws + o_X));

  // 10) decoder: H1 = X @ d_w1^T + d_b1 (fp32; pos region now dead)
  k_gemm<0><<<dim3(NTOT / 64, HIDN / 64), 128, 0, stream>>>(
      (const _Float16*)(ws + o_X), DD, IDBLK, 0,
      (const _Float16*)(ws + o_dw1), DD, d_b1, ws + o_H1, HIDN, DD);
  k_d1post<<<NTOT, HIDN, 0, stream>>>((const float*)(ws + o_H1), dg, dbe,
                                      (_Float16*)(ws + o_H1h));

  // 11) H2 = H1h @ d_w2^T + d_b2 (fp32)
  k_gemm<0><<<dim3(NTOT / 64, HIDN / 64), 128, 0, stream>>>(
      (const _Float16*)(ws + o_H1h), HIDN, IDBLK, 0,
      (const _Float16*)(ws + o_dw2), HIDN, d_b2, ws + o_H2, HIDN, HIDN);

  // 12) recon = GELU(H2) @ d_w3^T + d_b3
  k_final<<<NTOT / 128, 128, 0, stream>>>((const float*)(ws + o_H2), d_w3, d_b3, recon);

  (void)in_sizes; (void)n_in; (void)out_size; (void)ws_size;
}
